// NONA_15169824489967
// MI455X (gfx1250) — compile-verified, matrix operand order checked by reading, not run on hardware
//
#include <hip/hip_runtime.h>
#include <stdint.h>

// ---------------------------------------------------------------------------
// NONA forward, fused flash-attention style, for MI455X (gfx1250, wave32).
// out = softmax(-cdist(x, xn), axis=1) @ y   (global-max shift cancels)
// Staging uses GLOBAL_LOAD_ASYNC_TO_LDS_B128 (ASYNCcnt) double-buffering.
// ---------------------------------------------------------------------------

typedef unsigned short ushort_t;
typedef __bf16 bf16;
typedef __attribute__((ext_vector_type(8)))  bf16  bf16x8;
typedef __attribute__((ext_vector_type(16))) bf16  v16bf;
typedef __attribute__((ext_vector_type(8)))  float v8f;

constexpr int NQ = 4096;   // queries
constexpr int NK = 8192;   // neighbors
constexpr int DD = 1024;   // feature dim
constexpr int CC = 128;    // output classes

constexpr int BM = 128;    // query rows per workgroup
constexpr int BN = 128;    // neighbor cols per j-iteration
constexpr int BK = 64;     // K-dim staging depth (2 WMMA k-steps)
constexpr int KPAD = 8;
constexpr int KSTR = BK + KPAD;   // 72 elems -> 144B rows (16B aligned, bank-skewed)
constexpr int NSTR = BN + 8;      // 136 elems -> 272B rows (16B aligned, bank-skewed)
constexpr int NSPLIT = 8;         // neighbor-range split (flash-decoding style)
constexpr int JPER = (NK / BN) / NSPLIT;  // 8 j-blocks per split
constexpr int KTILES = DD / BK;           // 16 K-tiles per j-block

// LDS element offsets (dynamic LDS, no static __shared__ -> base offset 0)
constexpr int ASZ = BM * KSTR;            // 9216 elems / buffer
constexpr int BSZ = BN * KSTR;
constexpr int A_OFF[2] = {0, ASZ};
constexpr int B_OFF[2] = {2 * ASZ, 2 * ASZ + BSZ};
constexpr int Y_OFF = 2 * ASZ + 2 * BSZ;          // 36864 elems
constexpr int P_OFF = Y_OFF + CC * NSTR;          // 54272 elems
constexpr int LDS_ELEMS = P_OFF + BM * NSTR;      // 71680 elems = 143360 B

// f32 -> bf16 round-to-nearest-even
__device__ __forceinline__ ushort_t f2bf(float f) {
  union { float f; uint32_t u; } v; v.f = f;
  uint32_t u = v.u;
  uint32_t r = u + 0x7FFFu + ((u >> 16) & 1u);
  return (ushort_t)(r >> 16);
}

// 16-element bf16 fragment from LDS: elems 0..7 at p[0..7], 8..15 at p[16..23]
// (matches the documented 16-bit A/B VGPR layout: per-lane K runs
// {kb..kb+7} and {kb+16..kb+23}).
__device__ __forceinline__ v16bf ldsFrag(const ushort_t* p) {
  bf16x8 lo = *(const bf16x8*)(p);
  bf16x8 hi = *(const bf16x8*)(p + 16);
  return __builtin_shufflevector(lo, hi, 0,1,2,3,4,5,6,7,8,9,10,11,12,13,14,15);
}

// Async DMA: 16 bytes global -> LDS, no VGPR data, tracked by ASYNCcnt.
__device__ __forceinline__ void asyncCp16(uint32_t lds_byte_off, const void* g) {
  asm volatile("global_load_async_to_lds_b128 %0, %1, off"
               :: "v"(lds_byte_off), "v"((unsigned long long)(uintptr_t)g)
               : "memory");
}
__device__ __forceinline__ void waitAsync8() {
  asm volatile("s_wait_asynccnt 0x8" ::: "memory");
}
__device__ __forceinline__ void waitAsync0() {
  asm volatile("s_wait_asynccnt 0x0" ::: "memory");
}

// ---------------------------------------------------------------------------
// Prep: per-row sum of squares + f32 -> bf16 conversion. One wave per row.
// ---------------------------------------------------------------------------
__global__ void rowsq_cvt(const float* __restrict__ src,
                          ushort_t* __restrict__ dstb,
                          float* __restrict__ sq) {
  const int row  = blockIdx.x * 8 + (threadIdx.x >> 5);
  const int lane = threadIdx.x & 31;
  const float4* s4 = (const float4*)(src + (size_t)row * DD);
  uint2* d2 = (uint2*)(dstb + (size_t)row * DD);
  float ss = 0.f;
#pragma unroll
  for (int w = 0; w < DD / 128; ++w) {
    const int idx = w * 32 + lane;
    float4 v = s4[idx];
    ss += v.x * v.x + v.y * v.y + v.z * v.z + v.w * v.w;
    uint2 p;
    p.x = (uint32_t)f2bf(v.x) | ((uint32_t)f2bf(v.y) << 16);
    p.y = (uint32_t)f2bf(v.z) | ((uint32_t)f2bf(v.w) << 16);
    d2[idx] = p;
  }
#pragma unroll
  for (int off = 16; off >= 1; off >>= 1) ss += __shfl_xor(ss, off, 32);
  if (lane == 0) sq[row] = ss;
}

// ---------------------------------------------------------------------------
// Prep: y [NK][CC] f32 -> yT [CC][NK] bf16 (so GEMM2 B-frags are contiguous)
// ---------------------------------------------------------------------------
__global__ void ytrans(const float* __restrict__ y, ushort_t* __restrict__ yT) {
  const int gid = blockIdx.x * blockDim.x + threadIdx.x;
  const int c = gid & (CC - 1);
  const int k = gid >> 7;
  yT[(size_t)c * NK + k] = f2bf(y[(size_t)k * CC + c]);
}

// ---------------------------------------------------------------------------
// Main fused kernel. grid = (NQ/BM, NSPLIT), block = 256 (8 wave32s).
// ---------------------------------------------------------------------------
__global__ __launch_bounds__(256, 1)
void nona_flash(const ushort_t* __restrict__ xb, const ushort_t* __restrict__ nb,
                const ushort_t* __restrict__ yT, const float* __restrict__ asq,
                const float* __restrict__ bsq,
                float* __restrict__ opart, float* __restrict__ mpart,
                float* __restrict__ spart) {
  extern __shared__ __align__(16) ushort_t lds[];

  const int tid   = threadIdx.x;
  const int wave  = tid >> 5;
  const int lane  = tid & 31;
  const int lh    = lane >> 4;     // half-wave (C/D layout M-group)
  const int ln    = lane & 15;
  const int qbase = blockIdx.x * BM;
  const int split = blockIdx.y;
  const int jb0   = split * JPER;

  const int cpyRow = tid >> 1, cpyHalf = tid & 1;
  // per-thread constant pieces of the staging addresses
  const size_t gA0 = (size_t)(qbase + cpyRow) * DD + cpyHalf * 32;  // + kk + q*8
  const uint32_t lA0 = (uint32_t)(cpyRow * KSTR + cpyHalf * 32);    // elems, + buf base
  const uint32_t lB0 = lA0;

  // Issue one 64-deep K-tile (A:query, B:neighbor) into staging buffer `buf`.
  auto issueTile = [&](int nbase, int kk, int buf) {
#pragma unroll
    for (int q = 0; q < 4; ++q)
      asyncCp16((uint32_t)(A_OFF[buf] + lA0 + q * 8) * 2u,
                xb + gA0 + kk + q * 8);
    const size_t gB = (size_t)(nbase + cpyRow) * DD + cpyHalf * 32 + kk;
#pragma unroll
    for (int q = 0; q < 4; ++q)
      asyncCp16((uint32_t)(B_OFF[buf] + lB0 + q * 8) * 2u,
                nb + gB + q * 8);
  };

  float asql[8];
#pragma unroll
  for (int i = 0; i < 8; ++i) asql[i] = asq[qbase + wave * 16 + lh * 8 + i];

  float rowmax[8], rowsum[8];
  v8f O[8];
#pragma unroll
  for (int i = 0; i < 8; ++i) { rowmax[i] = -__builtin_inff(); rowsum[i] = 0.f; }
#pragma unroll
  for (int t = 0; t < 8; ++t)
#pragma unroll
    for (int i = 0; i < 8; ++i) O[t][i] = 0.f;

  // Prologue: first tile of first j-block in flight before anything else.
  issueTile(jb0 * BN, 0, 0);

  for (int jj = 0; jj < JPER; ++jj) {
    const int nbase = (jb0 + jj) * BN;

    // Stage yT tile [CC][BN] -> Yls (synchronous; 1/16 of the traffic).
    {
      const uint4* src = (const uint4*)(yT + (size_t)cpyRow * NK + nbase + cpyHalf * 64);
      uint4* dst = (uint4*)(lds + Y_OFF + cpyRow * NSTR + cpyHalf * 64);
#pragma unroll
      for (int q = 0; q < 8; ++q) dst[q] = src[q];
    }

    float bsql[8];
#pragma unroll
    for (int t = 0; t < 8; ++t) bsql[t] = bsq[nbase + t * 16 + ln];

    v8f S[8];
#pragma unroll
    for (int t = 0; t < 8; ++t)
#pragma unroll
      for (int i = 0; i < 8; ++i) S[t][i] = 0.f;

    // ---- GEMM1: S = x_block @ xn_block^T  (bf16 WMMA, f32 accum) ----
    // Async double-buffered pipeline on ASYNCcnt (8 ops per tile per thread).
    for (int it = 0; it < KTILES; ++it) {
      const int buf = it & 1;
      bool more = true;
      if (it + 1 < KTILES) {
        issueTile(nbase, (it + 1) * BK, (it + 1) & 1);
      } else if (jj + 1 < JPER) {
        issueTile(nbase + BN, 0, 0);     // next j-block's tile 0, under epilogue
      } else {
        more = false;
      }
      if (more) waitAsync8(); else waitAsync0();   // own tile `it` complete
      __syncthreads();                             // everyone's tile complete
#pragma unroll
      for (int ks = 0; ks < BK; ks += 32) {
        v16bf af = ldsFrag(lds + A_OFF[buf] + (wave * 16 + ln) * KSTR + ks + lh * 8);
#pragma unroll
        for (int t = 0; t < 8; ++t) {
          v16bf bfb = ldsFrag(lds + B_OFF[buf] + (t * 16 + ln) * KSTR + ks + lh * 8);
          S[t] = __builtin_amdgcn_wmma_f32_16x16x32_bf16(
              false, af, false, bfb, (short)0, S[t], false, false);
        }
      }
      __syncthreads();   // reads done before this buffer is re-filled
    }

    // ---- scores + online softmax (overlaps next j's tile-0 DMA) ----
    float mblk[8];
#pragma unroll
    for (int i = 0; i < 8; ++i) mblk[i] = -__builtin_inff();
#pragma unroll
    for (int t = 0; t < 8; ++t)
#pragma unroll
      for (int i = 0; i < 8; ++i) {
        float d2 = asql[i] + bsql[t] - 2.f * S[t][i];
        float sc = -sqrtf(fmaxf(d2, 0.f));
        S[t][i] = sc;
        mblk[i] = fmaxf(mblk[i], sc);
      }
#pragma unroll
    for (int i = 0; i < 8; ++i) {
#pragma unroll
      for (int off = 1; off < 16; off <<= 1)   // reduce across the 16-lane M-group
        mblk[i] = fmaxf(mblk[i], __shfl_xor(mblk[i], off, 32));
      float nm = fmaxf(rowmax[i], mblk[i]);
      float al = __expf(rowmax[i] - nm);
      rowmax[i] = nm;
      rowsum[i] *= al;
#pragma unroll
      for (int t = 0; t < 8; ++t) O[t][i] *= al;
    }
    float sblk[8];
#pragma unroll
    for (int i = 0; i < 8; ++i) sblk[i] = 0.f;
#pragma unroll
    for (int t = 0; t < 8; ++t)
#pragma unroll
      for (int i = 0; i < 8; ++i) {
        float p = __expf(S[t][i] - rowmax[i]);
        S[t][i] = p;
        sblk[i] += p;
      }
#pragma unroll
    for (int i = 0; i < 8; ++i) {
#pragma unroll
      for (int off = 1; off < 16; off <<= 1) sblk[i] += __shfl_xor(sblk[i], off, 32);
      rowsum[i] += sblk[i];
    }

    // P (C/D layout, f32) -> LDS (bf16, A-layout source)
#pragma unroll
    for (int t = 0; t < 8; ++t)
#pragma unroll
      for (int i = 0; i < 8; ++i)
        lds[P_OFF + (wave * 16 + lh * 8 + i) * NSTR + t * 16 + ln] = f2bf(S[t][i]);
    __syncthreads();

    // ---- GEMM2: O += P @ y_block ----
#pragma unroll
    for (int ks = 0; ks < BN; ks += 32) {
      v16bf pf = ldsFrag(lds + P_OFF + (wave * 16 + ln) * NSTR + ks + lh * 8);
#pragma unroll
      for (int t = 0; t < 8; ++t) {
        v16bf yf = ldsFrag(lds + Y_OFF + (t * 16 + ln) * NSTR + ks + lh * 8);
        O[t] = __builtin_amdgcn_wmma_f32_16x16x32_bf16(
            false, pf, false, yf, (short)0, O[t], false, false);
      }
    }
    __syncthreads();   // before next iteration restages Yls
  }

  // ---- emit split-partials (combined by nona_combine) ----
#pragma unroll
  for (int i = 0; i < 8; ++i) {
    const int row = qbase + wave * 16 + lh * 8 + i;
    const size_t base = (size_t)split * NQ + row;
    if (ln == 0) { mpart[base] = rowmax[i]; spart[base] = rowsum[i]; }
#pragma unroll
    for (int t = 0; t < 8; ++t)
      opart[base * CC + t * 16 + ln] = O[t][i];
  }
}

// ---------------------------------------------------------------------------
// Combine the NSPLIT partial softmax accumulators.
// ---------------------------------------------------------------------------
__global__ void nona_combine(const float* __restrict__ opart,
                             const float* __restrict__ mpart,
                             const float* __restrict__ spart,
                             float* __restrict__ out) {
  const int gid = blockIdx.x * blockDim.x + threadIdx.x;
  const int c = gid & (CC - 1);
  const int row = gid >> 7;
  float M = -__builtin_inff();
#pragma unroll
  for (int s = 0; s < NSPLIT; ++s) M = fmaxf(M, mpart[(size_t)s * NQ + row]);
  float tot = 0.f, acc = 0.f;
#pragma unroll
  for (int s = 0; s < NSPLIT; ++s) {
    const float w = __expf(mpart[(size_t)s * NQ + row] - M);
    tot += spart[(size_t)s * NQ + row] * w;
    acc += opart[((size_t)s * NQ + row) * CC + c] * w;
  }
  out[(size_t)row * CC + c] = acc / tot;
}

// ---------------------------------------------------------------------------
extern "C" void kernel_launch(void* const* d_in, const int* in_sizes, int n_in,
                              void* d_out, int out_size, void* d_ws, size_t ws_size,
                              hipStream_t stream) {
  (void)in_sizes; (void)n_in; (void)out_size; (void)ws_size;
  const float* x  = (const float*)d_in[0];
  const float* xn = (const float*)d_in[1];
  const float* y  = (const float*)d_in[2];
  float* out = (float*)d_out;

  char* ws = (char*)d_ws;
  size_t off = 0;
  ushort_t* xb = (ushort_t*)(ws + off); off += (size_t)NQ * DD * 2;   // 8 MB
  ushort_t* nb = (ushort_t*)(ws + off); off += (size_t)NK * DD * 2;   // 16 MB
  ushort_t* yT = (ushort_t*)(ws + off); off += (size_t)CC * NK * 2;   // 2 MB
  float* asq   = (float*)(ws + off);    off += (size_t)NQ * 4;
  float* bsq   = (float*)(ws + off);    off += (size_t)NK * 4;
  float* opart = (float*)(ws + off);    off += (size_t)NSPLIT * NQ * CC * 4; // 16 MB
  float* mpart = (float*)(ws + off);    off += (size_t)NSPLIT * NQ * 4;
  float* spart = (float*)(ws + off);    off += (size_t)NSPLIT * NQ * 4;

  rowsq_cvt<<<NQ / 8, 256, 0, stream>>>(x,  xb, asq);
  rowsq_cvt<<<NK / 8, 256, 0, stream>>>(xn, nb, bsq);
  ytrans<<<(NK * CC) / 256, 256, 0, stream>>>(y, yT);

  const size_t ldsBytes = (size_t)LDS_ELEMS * sizeof(ushort_t);   // ~140 KB
  dim3 grid(NQ / BM, NSPLIT);
  nona_flash<<<grid, 256, ldsBytes, stream>>>(xb, nb, yT, asq, bsq,
                                              opart, mpart, spart);

  nona_combine<<<(NQ * CC) / 256, 256, 0, stream>>>(opart, mpart, spart, out);
}
